// MultiHeadSelfAttention_30717606101351
// MI455X (gfx1250) — matrix operand order repaired and checked by reading, not compiled, not run on hardware
//
#include <hip/hip_runtime.h>
#include <hip/hip_bf16.h>
#include <math.h>

#define BB 4
#define NN 2048
#define DD 1024
#define HH 16
#define DH 64

typedef __bf16 bf16_t;
typedef bf16_t v16bf __attribute__((ext_vector_type(16)));
typedef bf16_t v8bf  __attribute__((ext_vector_type(8)));
typedef float  v8f   __attribute__((ext_vector_type(8)));

__device__ __forceinline__ v8f wmma_bf16f32(v16bf a, v16bf b, v8f c) {
  return __builtin_amdgcn_wmma_f32_16x16x32_bf16(false, a, false, b, (short)0, c,
                                                 false, false);
}

// 16-element bf16 fragment from two 16B chunks (chunk1 at +gap elements).
__device__ __forceinline__ v16bf ld_frag(const bf16_t* p, int gap) {
  union { v16bf v; v8bf h[2]; } u;
  u.h[0] = *reinterpret_cast<const v8bf*>(p);
  u.h[1] = *reinterpret_cast<const v8bf*>(p + gap);
  return u.v;
}

// 16 contiguous bf16 (32B, 32B-aligned): B-matrix fragment.
__device__ __forceinline__ v16bf ld_frag16(const bf16_t* p) {
  return *reinterpret_cast<const v16bf*>(p);
}

__device__ __forceinline__ v8f vzero8() {
  v8f z;
#pragma unroll
  for (int i = 0; i < 8; ++i) z[i] = 0.0f;
  return z;
}

// ---------------------------------------------------------------------------
// Kernel 0: bulk f32 -> bf16 conversion (x and weights), done once.
// ---------------------------------------------------------------------------
__global__ __launch_bounds__(256)
void cvt_bf16_kernel(const float* __restrict__ src, bf16_t* __restrict__ dst,
                     int n8) {
  const int i = blockIdx.x * 256 + threadIdx.x;
  if (i < n8) {
    v8f v = reinterpret_cast<const v8f*>(src)[i];
    v8bf o;
#pragma unroll
    for (int j = 0; j < 8; ++j) o[j] = (bf16_t)v[j];
    reinterpret_cast<v8bf*>(dst)[i] = o;
  }
}

// ---------------------------------------------------------------------------
// Kernel 1: Q/K/V projection GEMM, all-bf16 operands. One wave -> 32 rows x
// 64 features of ONE projection (w = blockIdx%3); B-fragments reused across
// both 16-row subtiles. Q scaled by 1/sqrt(Dh). K: [B,H,n,Dh]; V stored
// transposed [B,H,Dh,n] so attention B-fragments are contiguous.
// ---------------------------------------------------------------------------
__global__ __launch_bounds__(32)
void qkv_gemm_kernel(const bf16_t* __restrict__ xb,
                     const bf16_t* __restrict__ wqb,
                     const bf16_t* __restrict__ wkb,
                     const bf16_t* __restrict__ wvb,
                     const float* __restrict__ bq, const float* __restrict__ bk,
                     const float* __restrict__ bv,
                     bf16_t* __restrict__ qb, bf16_t* __restrict__ kb,
                     bf16_t* __restrict__ vtb) {
  const int lane = threadIdx.x & 31;
  const int ln = lane & 15, half = lane >> 4;
  int id = blockIdx.x;
  const int w = id % 3; id /= 3;
  const int fbase = (id & 15) * 64; id >>= 4;
  const int mt = id;  // 0..255, 32-row tiles

  const bf16_t* W = (w == 0) ? wqb : (w == 1) ? wkb : wvb;

  v8f acc[2][4];
#pragma unroll
  for (int mm = 0; mm < 2; ++mm)
#pragma unroll
    for (int t = 0; t < 4; ++t) acc[mm][t] = vzero8();

  const bf16_t* xr0 = xb + (size_t)(mt * 32 + ln) * DD;
  const bf16_t* xr1 = xr0 + (size_t)16 * DD;
  for (int ks = 0; ks < DD / 32; ++ks) {
    const int ko = ks * 32 + half * 16;
    v16bf a0 = ld_frag(xr0 + ko, 8);
    v16bf a1 = ld_frag(xr1 + ko, 8);
#pragma unroll
    for (int t = 0; t < 4; ++t) {
      v16bf bfr =
          ld_frag16(W + (size_t)(fbase + t * 16 + ln) * DD + ks * 32 + half * 16);
      acc[0][t] = wmma_bf16f32(a0, bfr, acc[0][t]);
      acc[1][t] = wmma_bf16f32(a1, bfr, acc[1][t]);
    }
  }

#pragma unroll
  for (int t = 0; t < 4; ++t) {
    const int f = fbase + t * 16 + ln;
    const int h = f >> 6, dh = f & 63;
    const float bias = (w == 0) ? bq[f] : (w == 1) ? bk[f] : bv[f];
#pragma unroll
    for (int mm = 0; mm < 2; ++mm) {
#pragma unroll
      for (int r = 0; r < 8; ++r) {
        const int m = mt * 32 + mm * 16 + r + half * 8;
        const int b = m >> 11, nq = m & (NN - 1);
        const size_t head = (size_t)b * HH + h;
        const float v = acc[mm][t][r] + bias;
        if (w == 0) {
          qb[(head * NN + nq) * DH + dh] = (bf16_t)(v * 0.125f);
        } else if (w == 1) {
          kb[(head * NN + nq) * DH + dh] = (bf16_t)v;
        } else {
          vtb[(head * DH + dh) * NN + nq] = (bf16_t)v;
        }
      }
    }
  }
}

// ---------------------------------------------------------------------------
// Kernel 2: flash attention, 32 queries per wave (2 x 16-row subtiles share
// every K/V fragment). Online softmax over 32-key blocks; P converted
// C-layout -> A-layout via 2KB LDS.
// ---------------------------------------------------------------------------
__global__ __launch_bounds__(32)
void attn_kernel(const float* __restrict__ bias, const bf16_t* __restrict__ qb,
                 const bf16_t* __restrict__ kbuf, const bf16_t* __restrict__ vtb,
                 bf16_t* __restrict__ ao) {
  const int lane = threadIdx.x & 31;
  const int ln = lane & 15, half = lane >> 4;
  int wg = blockIdx.x;
  const int qt = wg & 63; wg >>= 6;   // 64 tiles of 32 queries
  const int h = wg & 15;
  const int b = wg >> 4;

  const bf16_t* Q  = qb   + (((size_t)b * HH + h) * NN + qt * 32) * DH;
  const bf16_t* K  = kbuf +  ((size_t)b * HH + h) * NN * DH;
  const bf16_t* Vt = vtb  +  ((size_t)b * HH + h) * DH * NN;
  const float*  bs = bias +  ((size_t)b * NN + qt * 32) * NN;

  v16bf aQ[2][2];
#pragma unroll
  for (int mq = 0; mq < 2; ++mq)
#pragma unroll
    for (int s = 0; s < 2; ++s)
      aQ[mq][s] = ld_frag(Q + (size_t)(mq * 16 + ln) * DH + s * 32 + half * 16, 8);

  float mrow[2][8], lrow[2][8];
  v8f acc[2][4];
#pragma unroll
  for (int mq = 0; mq < 2; ++mq) {
#pragma unroll
    for (int r = 0; r < 8; ++r) { mrow[mq][r] = -1e30f; lrow[mq][r] = 0.0f; }
#pragma unroll
    for (int t = 0; t < 4; ++t) acc[mq][t] = vzero8();
  }

  __shared__ __align__(16) bf16_t Pl[32 * 32];

  for (int kblk = 0; kblk < NN / 32; ++kblk) {
    const int key0 = kblk * 32;

    // S = Q K^T + bias   (32 x 32 tile, 8 WMMAs; K fragments reused 2x)
    v8f S[2][2];
#pragma unroll
    for (int j = 0; j < 2; ++j) {
#pragma unroll
      for (int s = 0; s < 2; ++s) {
        v16bf bK =
            ld_frag16(K + (size_t)(key0 + j * 16 + ln) * DH + s * 32 + half * 16);
#pragma unroll
        for (int mq = 0; mq < 2; ++mq) {
          if (s == 0) S[mq][j] = vzero8();
          S[mq][j] = wmma_bf16f32(aQ[mq][s], bK, S[mq][j]);
        }
      }
#pragma unroll
      for (int mq = 0; mq < 2; ++mq) {
        const float* bp =
            bs + (size_t)(mq * 16 + half * 8) * NN + key0 + j * 16 + ln;
#pragma unroll
        for (int r = 0; r < 8; ++r) S[mq][j][r] += bp[(size_t)r * NN];
      }
    }

    // Online softmax per 16-row subtile; stage P into LDS (32x32 bf16).
#pragma unroll
    for (int mq = 0; mq < 2; ++mq) {
      float p0[8], p1[8], scale[8];
#pragma unroll
      for (int r = 0; r < 8; ++r) {
        float tm = fmaxf(S[mq][0][r], S[mq][1][r]);
#pragma unroll
        for (int msk = 1; msk < 16; msk <<= 1)
          tm = fmaxf(tm, __shfl_xor(tm, msk, 32));
        const float mn = fmaxf(mrow[mq][r], tm);
        scale[r] = __expf(mrow[mq][r] - mn);
        mrow[mq][r] = mn;
        p0[r] = __expf(S[mq][0][r] - mn);
        p1[r] = __expf(S[mq][1][r] - mn);
        float ls = p0[r] + p1[r];
#pragma unroll
        for (int msk = 1; msk < 16; msk <<= 1) ls += __shfl_xor(ls, msk, 32);
        lrow[mq][r] = lrow[mq][r] * scale[r] + ls;
      }
#pragma unroll
      for (int t = 0; t < 4; ++t)
#pragma unroll
        for (int r = 0; r < 8; ++r) acc[mq][t][r] *= scale[r];
#pragma unroll
      for (int r = 0; r < 8; ++r) {
        const int row = mq * 16 + r + half * 8;
        Pl[row * 32 + ln]      = (bf16_t)p0[r];
        Pl[row * 32 + 16 + ln] = (bf16_t)p1[r];
      }
    }
    __syncthreads();
    v16bf aP[2];
#pragma unroll
    for (int mq = 0; mq < 2; ++mq)
      aP[mq] = ld_frag(&Pl[(mq * 16 + ln) * 32 + half * 16], 8);
    __syncthreads();

    // O += P V   (8 WMMAs; V fragments reused 2x)
#pragma unroll
    for (int t = 0; t < 4; ++t) {
      v16bf bV = ld_frag16(Vt + (size_t)(t * 16 + ln) * NN + key0 + half * 16);
#pragma unroll
      for (int mq = 0; mq < 2; ++mq)
        acc[mq][t] = wmma_bf16f32(aP[mq], bV, acc[mq][t]);
    }
  }

  // Normalize and store to attn_out [B, N, D] (heads concatenated).
#pragma unroll
  for (int mq = 0; mq < 2; ++mq) {
    float inv[8];
#pragma unroll
    for (int r = 0; r < 8; ++r) inv[r] = 1.0f / lrow[mq][r];
#pragma unroll
    for (int t = 0; t < 4; ++t) {
      const int col = h * DH + t * 16 + ln;
#pragma unroll
      for (int r = 0; r < 8; ++r) {
        const int q = qt * 32 + mq * 16 + r + half * 8;
        ao[((size_t)b * NN + q) * DD + col] = (bf16_t)(acc[mq][t][r] * inv[r]);
      }
    }
  }
}

// ---------------------------------------------------------------------------
// Kernel 3: output projection + bias + residual (M=32 per wave, bf16 weights).
// ---------------------------------------------------------------------------
__global__ __launch_bounds__(32)
void oproj_kernel(const bf16_t* __restrict__ ao, const bf16_t* __restrict__ wob,
                  const float* __restrict__ bo, const float* __restrict__ x,
                  float* __restrict__ hbuf) {
  const int lane = threadIdx.x & 31;
  const int ln = lane & 15, half = lane >> 4;
  const int mt = blockIdx.x >> 4;  // 0..255, 32-row tiles
  const int fbase = (blockIdx.x & 15) * 64;

  v8f acc[2][4];
#pragma unroll
  for (int mm = 0; mm < 2; ++mm)
#pragma unroll
    for (int t = 0; t < 4; ++t) acc[mm][t] = vzero8();

  const bf16_t* ar0 = ao + (size_t)(mt * 32 + ln) * DD;
  const bf16_t* ar1 = ar0 + (size_t)16 * DD;
  for (int ks = 0; ks < DD / 32; ++ks) {
    const int ko = ks * 32 + half * 16;
    v16bf a0 = ld_frag(ar0 + ko, 8);
    v16bf a1 = ld_frag(ar1 + ko, 8);
#pragma unroll
    for (int t = 0; t < 4; ++t) {
      v16bf bfr = ld_frag16(wob + (size_t)(fbase + t * 16 + ln) * DD + ks * 32 +
                            half * 16);
      acc[0][t] = wmma_bf16f32(a0, bfr, acc[0][t]);
      acc[1][t] = wmma_bf16f32(a1, bfr, acc[1][t]);
    }
  }
#pragma unroll
  for (int t = 0; t < 4; ++t) {
    const int f = fbase + t * 16 + ln;
    const float bov = bo[f];
#pragma unroll
    for (int mm = 0; mm < 2; ++mm) {
#pragma unroll
      for (int r = 0; r < 8; ++r) {
        const size_t m = (size_t)mt * 32 + mm * 16 + r + half * 8;
        hbuf[m * DD + f] = acc[mm][t][r] + bov + x[m * DD + f];
      }
    }
  }
}

// ---------------------------------------------------------------------------
// Kernel 4: LayerNorm over d=1024 per row. 256 threads, float4 per thread.
// ---------------------------------------------------------------------------
__global__ __launch_bounds__(256)
void ln_kernel(const float* __restrict__ hbuf, const float* __restrict__ gamma,
               const float* __restrict__ beta, float* __restrict__ out) {
  const int row = blockIdx.x, tid = threadIdx.x;
  const float4 v = reinterpret_cast<const float4*>(hbuf + (size_t)row * DD)[tid];
  float s = v.x + v.y + v.z + v.w;
  float s2 = v.x * v.x + v.y * v.y + v.z * v.z + v.w * v.w;
#pragma unroll
  for (int m = 1; m < 32; m <<= 1) {
    s  += __shfl_xor(s, m, 32);
    s2 += __shfl_xor(s2, m, 32);
  }
  __shared__ float w1[8], w2[8];
  __shared__ float mu_s, rs_s;
  const int wv = tid >> 5;
  if ((tid & 31) == 0) { w1[wv] = s; w2[wv] = s2; }
  __syncthreads();
  if (tid == 0) {
    float S = 0.0f, S2 = 0.0f;
#pragma unroll
    for (int i = 0; i < 8; ++i) { S += w1[i]; S2 += w2[i]; }
    const float mu = S * (1.0f / DD);
    const float var = S2 * (1.0f / DD) - mu * mu;
    mu_s = mu;
    rs_s = rsqrtf(var + 1e-5f);
  }
  __syncthreads();
  const float mu = mu_s, rs = rs_s;
  const float4 g  = reinterpret_cast<const float4*>(gamma)[tid];
  const float4 bt = reinterpret_cast<const float4*>(beta)[tid];
  float4 o;
  o.x = (v.x - mu) * rs * g.x + bt.x;
  o.y = (v.y - mu) * rs * g.y + bt.y;
  o.z = (v.z - mu) * rs * g.z + bt.z;
  o.w = (v.w - mu) * rs * g.w + bt.w;
  reinterpret_cast<float4*>(out + (size_t)row * DD)[tid] = o;
}

extern "C" void kernel_launch(void* const* d_in, const int* in_sizes, int n_in,
                              void* d_out, int out_size, void* d_ws,
                              size_t ws_size, hipStream_t stream) {
  const float* x     = (const float*)d_in[0];
  const float* bias  = (const float*)d_in[1];
  const float* Wq    = (const float*)d_in[2];
  const float* bq    = (const float*)d_in[3];
  const float* Wk    = (const float*)d_in[4];
  const float* bk    = (const float*)d_in[5];
  const float* Wv    = (const float*)d_in[6];
  const float* bv    = (const float*)d_in[7];
  const float* Wo    = (const float*)d_in[8];
  const float* bo    = (const float*)d_in[9];
  const float* gamma = (const float*)d_in[10];
  const float* beta  = (const float*)d_in[11];
  float* out = (float*)d_out;

  char* ws = (char*)d_ws;
  const size_t qkvB = (size_t)BB * HH * NN * DH * 2;  // 16 MiB per tensor
  const size_t wB   = (size_t)DD * DD * 2;            // 2 MiB per weight
  size_t off = 0;
  bf16_t* qb   = (bf16_t*)(ws + off); off += qkvB;
  bf16_t* kb   = (bf16_t*)(ws + off); off += qkvB;
  bf16_t* vtb  = (bf16_t*)(ws + off); off += qkvB;
  bf16_t* ao   = (bf16_t*)(ws + off); off += qkvB;
  bf16_t* xbf  = (bf16_t*)(ws + off); off += qkvB;   // x: 8192x1024 bf16
  bf16_t* wqb  = (bf16_t*)(ws + off); off += wB;
  bf16_t* wkb  = (bf16_t*)(ws + off); off += wB;
  bf16_t* wvb  = (bf16_t*)(ws + off); off += wB;
  bf16_t* wob  = (bf16_t*)(ws + off); off += wB;
  float*  hbuf = (float*)(ws + off);

  const int xN8 = (int)((size_t)BB * NN * DD / 8);  // 1,048,576
  const int wN8 = (int)((size_t)DD * DD / 8);       // 131,072
  cvt_bf16_kernel<<<dim3((xN8 + 255) / 256), dim3(256), 0, stream>>>(x, xbf, xN8);
  cvt_bf16_kernel<<<dim3((wN8 + 255) / 256), dim3(256), 0, stream>>>(Wq, wqb, wN8);
  cvt_bf16_kernel<<<dim3((wN8 + 255) / 256), dim3(256), 0, stream>>>(Wk, wkb, wN8);
  cvt_bf16_kernel<<<dim3((wN8 + 255) / 256), dim3(256), 0, stream>>>(Wv, wvb, wN8);
  cvt_bf16_kernel<<<dim3((wN8 + 255) / 256), dim3(256), 0, stream>>>(Wo, wob, wN8);

  qkv_gemm_kernel<<<dim3(256 * 16 * 3), dim3(32), 0, stream>>>(
      xbf, wqb, wkb, wvb, bq, bk, bv, qb, kb, vtb);
  attn_kernel<<<dim3(BB * HH * (NN / 32)), dim3(32), 0, stream>>>(bias, qb, kb,
                                                                  vtb, ao);
  oproj_kernel<<<dim3(256 * 16), dim3(32), 0, stream>>>(ao, wob, bo, x, hbuf);
  ln_kernel<<<dim3(BB * NN), dim3(256), 0, stream>>>(hbuf, gamma, beta, out);
}
